// CausalSelfAttention_86698209837549
// MI455X (gfx1250) — matrix operand order repaired and checked by reading, not compiled, not run on hardware
//
#include <hip/hip_runtime.h>

#define T_SEQ 4096
#define C_EMB 768
#define NH    12
#define HD    64
#define N_QKV 2304

typedef __attribute__((ext_vector_type(16))) __bf16 v16bf;
typedef __attribute__((ext_vector_type(8)))  __bf16 v8bf;
typedef __attribute__((ext_vector_type(8)))  float  v8f;

// ---- WMMA wrapper: D = A(16x32 bf16) * B(32x16 bf16) + C(16x16 f32) ----
__device__ __forceinline__ v8f wmma_bf16(v16bf a, v16bf b, v8f c) {
  return __builtin_amdgcn_wmma_f32_16x16x32_bf16(
      /*neg_a=*/false, a, /*neg_b=*/false, b,
      /*c_mod=*/(short)0, c, /*reuse_a=*/false, /*reuse_b=*/false);
}

// ---- A-fragment loader (16x32, 16-bit): lane m=L%16 holds row m,
//      k = kb..kb+15 contiguous, kb = 16*(L>=16). `row` points at k=0. ----
__device__ __forceinline__ v16bf load_a16(const __bf16* row) {
  const int kb = (threadIdx.x & 16) ? 16 : 0;
  v8bf lo = *(const v8bf*)(row + kb);
  v8bf hi = *(const v8bf*)(row + kb + 8);
  v16bf f;
#pragma unroll
  for (int i = 0; i < 8; ++i) { f[i] = lo[i]; f[i + 8] = hi[i]; }
  return f;
}

// ---- B-fragment loader (32x16, 16-bit): lane n=L%16 holds col n,
//      k = kb..kb+15 contiguous, kb = 16*(L>=16). `row` = col-major col start. ----
__device__ __forceinline__ v16bf load_b16(const __bf16* row) {
  const int kb = (threadIdx.x & 16) ? 16 : 0;
  v8bf lo = *(const v8bf*)(row + kb);
  v8bf hi = *(const v8bf*)(row + kb + 8);
  v16bf f;
#pragma unroll
  for (int i = 0; i < 8; ++i) { f[i] = lo[i]; f[i + 8] = hi[i]; }
  return f;
}

// ===================== conversion / transpose kernels =====================
__global__ void k_cvt_bf16(const float* __restrict__ src, __bf16* __restrict__ dst, int n) {
  int i = blockIdx.x * blockDim.x + threadIdx.x;
  if (i < n) dst[i] = (__bf16)src[i];
}

// src[K][N] (fp32) -> dst[N][K] (bf16)
__global__ void k_tr_bf16(const float* __restrict__ src, __bf16* __restrict__ dst, int K, int N) {
  int i = blockIdx.x * blockDim.x + threadIdx.x;
  if (i < K * N) {
    int n = i / K, k = i % K;
    dst[i] = (__bf16)src[k * N + n];
  }
}

// ===================== QKV projection: qkv = Xb @ Wattn + b =====================
// One wave computes a 32(M) x 64(N) tile; B fragments reused across 2 M-subtiles.
// Stores Q,K as [H][T][D], V as [H][D][T] (bf16).
__global__ __launch_bounds__(128) void k_gemm_qkv(
    const __bf16* __restrict__ Xb, const __bf16* __restrict__ Wat,
    const float* __restrict__ bias,
    __bf16* __restrict__ Qb, __bf16* __restrict__ Kb, __bf16* __restrict__ Vtb) {
  const int NT = N_QKV / 64;  // 36 n-tiles
  int wid = blockIdx.x * (blockDim.x >> 5) + (threadIdx.x >> 5);
  int mt = wid / NT, nt = wid % NT;
  int t0 = mt * 32, n0 = nt * 64;
  int lane = threadIdx.x & 31, col = lane & 15, hi8 = (lane & 16) ? 8 : 0;

  v8f acc[2][4] = {};
  const __bf16* arow0 = Xb + (size_t)(t0 + col) * C_EMB;
  const __bf16* arow1 = Xb + (size_t)(t0 + 16 + col) * C_EMB;
  for (int k0 = 0; k0 < C_EMB; k0 += 32) {
    v16bf a0 = load_a16(arow0 + k0);
    v16bf a1 = load_a16(arow1 + k0);
#pragma unroll
    for (int s = 0; s < 4; ++s) {
      v16bf b = load_b16(Wat + (size_t)(n0 + s * 16 + col) * C_EMB + k0);
      acc[0][s] = wmma_bf16(a0, b, acc[0][s]);
      acc[1][s] = wmma_bf16(a1, b, acc[1][s]);
    }
  }
#pragma unroll
  for (int s = 0; s < 4; ++s) {
    int n = n0 + s * 16 + col;
    float bv = bias[n];
    int which = n / C_EMB, rem = n % C_EMB, h = rem / HD, d = rem % HD;
#pragma unroll
    for (int ms = 0; ms < 2; ++ms)
#pragma unroll
      for (int r = 0; r < 8; ++r) {
        int t = t0 + ms * 16 + r + hi8;
        float v = acc[ms][s][r] + bv;
        if (which == 0)      Qb[((size_t)h * T_SEQ + t) * HD + d]  = (__bf16)v;
        else if (which == 1) Kb[((size_t)h * T_SEQ + t) * HD + d]  = (__bf16)v;
        else                 Vtb[((size_t)h * HD + d) * T_SEQ + t] = (__bf16)v;
      }
  }
}

// ===================== flash attention (transposed tiles) =====================
// One wave: 32 queries (two 16-query subtiles) of one head.
// St = K*Q^T (keys x queries), O^T = V^T * P^T; K and V fragments shared by both subtiles.
__global__ __launch_bounds__(128) void k_attn(
    const __bf16* __restrict__ Qb, const __bf16* __restrict__ Kb,
    const __bf16* __restrict__ Vtb, __bf16* __restrict__ Yb) {
  const int QT = T_SEQ / 32;  // 128 query tiles of 32
  int wid = blockIdx.x * (blockDim.x >> 5) + (threadIdx.x >> 5);
  int h = wid / QT, qt = wid % QT;
  int q0 = qt * 32;
  int lane = threadIdx.x & 31, col = lane & 15, hi8 = (lane & 16) ? 8 : 0;
  bool upper = (lane & 16) != 0;

  const __bf16* Qh  = Qb  + (size_t)h * T_SEQ * HD;
  const __bf16* Kh  = Kb  + (size_t)h * T_SEQ * HD;
  const __bf16* Vth = Vtb + (size_t)h * HD * T_SEQ;

  // Q^T B-fragments: [subtile][d-chunk], loaded once
  v16bf qf[2][2];
#pragma unroll
  for (int sub = 0; sub < 2; ++sub) {
    const __bf16* qrow = Qh + (size_t)(q0 + sub * 16 + col) * HD;
    qf[sub][0] = load_b16(qrow);
    qf[sub][1] = load_b16(qrow + 32);
  }

  v8f acc[2][4] = {};                 // O^T per subtile: 64 x 16 f32
  float m[2] = {-1e30f, -1e30f}, l[2] = {0.f, 0.f};
  int qg[2] = {q0 + col, q0 + 16 + col};
  int ntiles = qt + 1;  // causal: key tiles of 32 up to the diagonal 32-block

  for (int j = 0; j < ntiles; ++j) {
    int kt0 = j * 32;
    const __bf16* k0row = Kh + (size_t)(kt0 + col) * HD;
    const __bf16* k1row = Kh + (size_t)(kt0 + 16 + col) * HD;
    // K A-fragments, shared by both query subtiles
    v16bf ka0 = load_a16(k0row);
    v16bf ka1 = load_a16(k0row + 32);
    v16bf kb0 = load_a16(k1row);
    v16bf kb1 = load_a16(k1row + 32);

    v16bf pt[2];
#pragma unroll
    for (int sub = 0; sub < 2; ++sub) {
      v8f st0 = v8f{}, st1 = v8f{};
      st0 = wmma_bf16(ka0, qf[sub][0], st0);
      st0 = wmma_bf16(ka1, qf[sub][1], st0);
      st1 = wmma_bf16(kb0, qf[sub][0], st1);
      st1 = wmma_bf16(kb1, qf[sub][1], st1);

      float s0[8], s1[8], mx = -1e30f;
#pragma unroll
      for (int r = 0; r < 8; ++r) {
        int kg0 = kt0 + r + hi8, kg1 = kg0 + 16;
        s0[r] = (kg0 <= qg[sub]) ? st0[r] * 0.125f : -1e30f;
        s1[r] = (kg1 <= qg[sub]) ? st1[r] * 0.125f : -1e30f;
        mx = fmaxf(mx, fmaxf(s0[r], s1[r]));
      }
      mx = fmaxf(mx, __shfl_xor(mx, 16, 32));  // per-query max over 32 keys
      float mn = fmaxf(m[sub], mx);
      float alpha = __expf(m[sub] - mn);
      float p0[8], p1[8], sum = 0.f;
#pragma unroll
      for (int r = 0; r < 8; ++r) {
        p0[r] = __expf(s0[r] - mn);
        p1[r] = __expf(s1[r] - mn);
        sum += p0[r] + p1[r];
      }
      sum += __shfl_xor(sum, 16, 32);
      l[sub] = l[sub] * alpha + sum;
      m[sub] = mn;
#pragma unroll
      for (int f = 0; f < 4; ++f)
#pragma unroll
        for (int r = 0; r < 8; ++r) acc[sub][f][r] *= alpha;

      // Repack P^T (C-layout) into B-fragment layout: lower half needs keys 0..15,
      // upper half keys 16..31 -> one cross-half shuffle per element pair.
#pragma unroll
      for (int r = 0; r < 8; ++r) {
        float o0 = __shfl_xor(p0[r], 16, 32);
        float o1 = __shfl_xor(p1[r], 16, 32);
        pt[sub][r]     = (__bf16)(upper ? o1    : p0[r]);
        pt[sub][r + 8] = (__bf16)(upper ? p1[r] : o0);
      }
    }
    // O^T += V^T * P^T : each V fragment feeds both subtiles
#pragma unroll
    for (int f = 0; f < 4; ++f) {
      v16bf vt = load_a16(Vth + (size_t)(f * 16 + col) * T_SEQ + kt0);
      acc[0][f] = wmma_bf16(vt, pt[0], acc[0][f]);
      acc[1][f] = wmma_bf16(vt, pt[1], acc[1][f]);
    }
  }

  // Epilogue: normalize and store bf16 pairs (consecutive channels) as b32
#pragma unroll
  for (int sub = 0; sub < 2; ++sub) {
    float inv = 1.0f / l[sub];
    int t = q0 + sub * 16 + col;
#pragma unroll
    for (int f = 0; f < 4; ++f)
#pragma unroll
      for (int r = 0; r < 8; r += 2) {
        int c = h * HD + f * 16 + r + hi8;
        union { __bf16 b[2]; unsigned u; } pk;
        pk.b[0] = (__bf16)(acc[sub][f][r] * inv);
        pk.b[1] = (__bf16)(acc[sub][f][r + 1] * inv);
        *(unsigned*)(Yb + (size_t)t * C_EMB + c) = pk.u;
      }
  }
}

// ===================== output projection: out = Yb @ Wproj + b =====================
__global__ __launch_bounds__(128) void k_gemm_proj(
    const __bf16* __restrict__ Yb, const __bf16* __restrict__ Wpt,
    const float* __restrict__ bias, float* __restrict__ out) {
  const int NT = C_EMB / 64;  // 12
  int wid = blockIdx.x * (blockDim.x >> 5) + (threadIdx.x >> 5);
  int mt = wid / NT, nt = wid % NT;
  int t0 = mt * 32, n0 = nt * 64;
  int lane = threadIdx.x & 31, col = lane & 15, hi8 = (lane & 16) ? 8 : 0;

  v8f acc[2][4] = {};
  const __bf16* arow0 = Yb + (size_t)(t0 + col) * C_EMB;
  const __bf16* arow1 = Yb + (size_t)(t0 + 16 + col) * C_EMB;
  for (int k0 = 0; k0 < C_EMB; k0 += 32) {
    v16bf a0 = load_a16(arow0 + k0);
    v16bf a1 = load_a16(arow1 + k0);
#pragma unroll
    for (int s = 0; s < 4; ++s) {
      v16bf b = load_b16(Wpt + (size_t)(n0 + s * 16 + col) * C_EMB + k0);
      acc[0][s] = wmma_bf16(a0, b, acc[0][s]);
      acc[1][s] = wmma_bf16(a1, b, acc[1][s]);
    }
  }
#pragma unroll
  for (int s = 0; s < 4; ++s) {
    int n = n0 + s * 16 + col;
    float bv = bias[n];
#pragma unroll
    for (int ms = 0; ms < 2; ++ms)
#pragma unroll
      for (int r = 0; r < 8; ++r) {
        int t = t0 + ms * 16 + r + hi8;
        out[(size_t)t * C_EMB + n] = acc[ms][s][r] + bv;
      }
  }
}

// ===================== host launcher =====================
extern "C" void kernel_launch(void* const* d_in, const int* in_sizes, int n_in,
                              void* d_out, int out_size, void* d_ws, size_t ws_size,
                              hipStream_t stream) {
  (void)in_sizes; (void)n_in; (void)out_size; (void)ws_size;
  const float* x      = (const float*)d_in[0];
  const float* w_attn = (const float*)d_in[1];
  const float* b_attn = (const float*)d_in[2];
  const float* w_proj = (const float*)d_in[3];
  const float* b_proj = (const float*)d_in[4];
  float* out = (float*)d_out;

  char* ws = (char*)d_ws;
  size_t off = 0;
  auto carve = [&](size_t bytes) {
    void* p = ws + off;
    off += (bytes + 255) & ~(size_t)255;
    return p;
  };
  __bf16* Xb  = (__bf16*)carve((size_t)T_SEQ * C_EMB * 2);
  __bf16* Wat = (__bf16*)carve((size_t)N_QKV * C_EMB * 2);   // [3C][C] transposed
  __bf16* Wpt = (__bf16*)carve((size_t)C_EMB * C_EMB * 2);   // [C][C] transposed
  __bf16* Qb  = (__bf16*)carve((size_t)NH * T_SEQ * HD * 2); // [H][T][D]
  __bf16* Kb  = (__bf16*)carve((size_t)NH * T_SEQ * HD * 2); // [H][T][D]
  __bf16* Vtb = (__bf16*)carve((size_t)NH * HD * T_SEQ * 2); // [H][D][T]
  __bf16* Yb  = (__bf16*)carve((size_t)T_SEQ * C_EMB * 2);   // attn output bf16

  int n;
  n = T_SEQ * C_EMB;
  k_cvt_bf16<<<(n + 255) / 256, 256, 0, stream>>>(x, Xb, n);
  n = C_EMB * N_QKV;
  k_tr_bf16<<<(n + 255) / 256, 256, 0, stream>>>(w_attn, Wat, C_EMB, N_QKV);
  n = C_EMB * C_EMB;
  k_tr_bf16<<<(n + 255) / 256, 256, 0, stream>>>(w_proj, Wpt, C_EMB, C_EMB);

  // QKV gemm: (4096/32) * (2304/64) = 4608 waves, 4 waves/block
  k_gemm_qkv<<<4608 / 4, 128, 0, stream>>>(Xb, Wat, b_attn, Qb, Kb, Vtb);
  // attention: 12 heads * 128 query tiles (32 wide) = 1536 waves
  k_attn<<<1536 / 4, 128, 0, stream>>>(Qb, Kb, Vtb, Yb);
  // projection: (4096/32) * (768/64) = 1536 waves
  k_gemm_proj<<<1536 / 4, 128, 0, stream>>>(Yb, Wpt, b_proj, out);
}
